// SimpleRNN_37984690766360
// MI455X (gfx1250) — matrix operand order repaired and checked by reading, not compiled, not run on hardware
//
#include <hip/hip_runtime.h>
#include <hip/hip_bf16.h>
#include <math.h>

// Tensor Data Mover preload path in rnn_scan: this toolchain exposes the
// 6-arg clang-23 form (u32x4, i32x8, i32x4, i32x4, i32x8, i32 cpol).
#define RNN_USE_TDM 1

// Problem dims (fixed by the reference)
#define RNN_B 32
#define RNN_S 2048
#define RNN_H 512
#define RNN_I 512
#define RNN_O 512

typedef __attribute__((ext_vector_type(16))) __bf16 bf16x16;
typedef __attribute__((ext_vector_type(8)))  __bf16 bf16x8;
typedef __attribute__((ext_vector_type(8)))  float  f32x8;
typedef __attribute__((ext_vector_type(4)))  unsigned short u16x4;
typedef __attribute__((ext_vector_type(8)))  unsigned short u16x8;
typedef __attribute__((ext_vector_type(4)))  unsigned int   u32x4;
typedef __attribute__((ext_vector_type(4)))  int            i32x4;
typedef __attribute__((ext_vector_type(8)))  int            i32x8;

// round-to-nearest-even f32 -> bf16
static __device__ __forceinline__ unsigned short f2bf(float f) {
  union { float f; unsigned u; } v; v.f = f;
  unsigned r = v.u + 0x7FFFu + ((v.u >> 16) & 1u);
  return (unsigned short)(r >> 16);
}

static __device__ __forceinline__ bf16x16 cat8(bf16x8 lo, bf16x8 hi) {
  return __builtin_shufflevector(lo, hi, 0,1,2,3,4,5,6,7,8,9,10,11,12,13,14,15);
}

// 16-bit A fragment (16x32, MxK). Lane L: M = L%16; K chunks at kb+8*(L/16)
// and kb+8*(L/16)+16, each 8 contiguous bf16 (one b128 load).
static __device__ __forceinline__ bf16x16
load_a_frag(const unsigned short* base, int ld, int m0, int kb) {
  const int lane = threadIdx.x & 31;
  const unsigned short* p = base + (size_t)(m0 + (lane & 15)) * ld + kb + ((lane >> 4) << 3);
  bf16x8 lo = *(const bf16x8*)(p);
  bf16x8 hi = *(const bf16x8*)(p + 16);
  return cat8(lo, hi);
}

// 16-bit B fragment (32x16, KxN) from N-major storage (row n holds K contiguous).
// Lane L: N = L%16; K chunks at kb+8*(L/16) and kb+8*(L/16)+16, 8 contiguous each.
static __device__ __forceinline__ bf16x16
load_b_frag(const unsigned short* base, int ld, int n0, int kb) {
  const int lane = threadIdx.x & 31;
  const unsigned short* p = base + (size_t)(n0 + (lane & 15)) * ld + kb + ((lane >> 4) << 3);
  bf16x8 lo = *(const bf16x8*)(p);
  bf16x8 hi = *(const bf16x8*)(p + 16);
  return cat8(lo, hi);
}

// C/D f32 16x16 store: lane L holds rows M = v + 8*(L/16), col N = L%16.
// bias is a dense per-column vector (never null) -> branch-free epilogue.
static __device__ __forceinline__ void
store_ctile_bias(float* __restrict__ C, int ldc, int m0, int n0, f32x8 acc,
                 const float* __restrict__ bias) {
  const int lane = threadIdx.x & 31;
  const int n = n0 + (lane & 15);
  const float b = bias[n];
  const int mb = m0 + ((lane >> 4) << 3);
#pragma unroll
  for (int v = 0; v < 8; ++v) C[(size_t)(mb + v) * ldc + n] = acc[v] + b;
}

// ---------------------------------------------------------------------------
// Kernel 0a: zero h0 + grid-barrier counter (graph-replay safe re-init)
__global__ void init_state(unsigned short* h0, int* bar) {
  int i = blockIdx.x * blockDim.x + threadIdx.x;
  if (i == 0) *bar = 0;
  if (i < RNN_B * RNN_H) h0[i] = 0;  // bf16 +0.0
}

// Kernel 0b: f32 -> bf16 weight conversion
__global__ __launch_bounds__(256) void cvt_bf16(const float* __restrict__ s,
                                                unsigned short* __restrict__ d, int n) {
  int i = (blockIdx.x * 256 + threadIdx.x) * 4;
  if (i + 3 < n) {
    float4 v = *(const float4*)(s + i);
    u16x4 r = { f2bf(v.x), f2bf(v.y), f2bf(v.z), f2bf(v.w) };
    *(u16x4*)(d + i) = r;
  }
}

// Kernel 0c: fold the three bias vectors once
__global__ void bias_prep(const float* __restrict__ a, const float* __restrict__ b,
                          const float* __restrict__ c, float* __restrict__ d) {
  int i = blockIdx.x * blockDim.x + threadIdx.x;
  if (i < RNN_H) d[i] = a[i] + b[i] + c[i];
}

// ---------------------------------------------------------------------------
// Kernel 1: xproj[m,n] = sum_k x[m,k]*Wxh[n,k] + bias_sum[n]
// M=65536, N=512, K=512. Block tile 128x64, 8 waves (4Mx2N), wave tile 32x32,
// BK=32. Double-buffered LDS: global loads for tile k+1 are in flight across
// the WMMAs on tile k (one barrier per iteration).
__global__ __launch_bounds__(256) void xproj_gemm(
    const float* __restrict__ x, const unsigned short* __restrict__ Wbf,
    const float* __restrict__ bias, float* __restrict__ xproj) {
  __shared__ unsigned short As[2][128 * 40];  // 32 bf16 + 8 pad per row
  __shared__ unsigned short Bs[2][64 * 40];

  const int tid = threadIdx.x;
  const int bm = blockIdx.y;   // 0..511
  const int bn = blockIdx.x;   // 0..7
  const int w  = tid >> 5;
  const int wm = w & 3;        // M sub-tile 0..3
  const int wn = w >> 2;       // N sub-tile 0..1

  f32x8 acc00 = {}, acc01 = {}, acc10 = {}, acc11 = {};

  const int arow = tid >> 1;         // 0..127
  const int acol = (tid & 1) << 4;   // 0 or 16 (f32 elems)
  const int brow = tid >> 2;         // 0..63
  const int bcol = (tid & 3) << 3;   // 0,8,16,24 (bf16 elems)

  const float*          asrc = x   + (size_t)(bm * 128 + arow) * RNN_I + acol;
  const unsigned short* bsrc = Wbf + (size_t)(bn * 64 + brow) * RNN_I + bcol;

  // Stage tile 0 into buffer 0
  {
    unsigned short tmp[16];
#pragma unroll
    for (int i = 0; i < 16; ++i) tmp[i] = f2bf(asrc[i]);
    u16x8* dst = (u16x8*)&As[0][arow * 40 + acol];
    dst[0] = *(const u16x8*)&tmp[0];
    dst[1] = *(const u16x8*)&tmp[8];
    *(u16x8*)&Bs[0][brow * 40 + bcol] = *(const u16x8*)bsrc;
  }
  __syncthreads();

  for (int kt = 0; kt < RNN_I / 32; ++kt) {
    const int cur = kt & 1;
    const int nxt = cur ^ 1;

    // Issue global loads for tile kt+1 (in flight during the WMMAs below)
    float4 areg0, areg1, areg2, areg3;
    u16x8 breg;
    if (kt < RNN_I / 32 - 1) {
      const float* sa = asrc + (kt + 1) * 32;
      areg0 = ((const float4*)sa)[0];
      areg1 = ((const float4*)sa)[1];
      areg2 = ((const float4*)sa)[2];
      areg3 = ((const float4*)sa)[3];
      breg  = *(const u16x8*)(bsrc + (kt + 1) * 32);
    }

    // Compute on the current buffer
    bf16x16 a0 = load_a_frag(As[cur], 40, wm * 32, 0);
    bf16x16 a1 = load_a_frag(As[cur], 40, wm * 32 + 16, 0);
    bf16x16 b0 = load_b_frag(Bs[cur], 40, wn * 32, 0);
    bf16x16 b1 = load_b_frag(Bs[cur], 40, wn * 32 + 16, 0);
    acc00 = __builtin_amdgcn_wmma_f32_16x16x32_bf16(false, a0, false, b0, (short)0, acc00, false, false);
    acc01 = __builtin_amdgcn_wmma_f32_16x16x32_bf16(false, a0, false, b1, (short)0, acc01, false, false);
    acc10 = __builtin_amdgcn_wmma_f32_16x16x32_bf16(false, a1, false, b0, (short)0, acc10, false, false);
    acc11 = __builtin_amdgcn_wmma_f32_16x16x32_bf16(false, a1, false, b1, (short)0, acc11, false, false);

    // Convert + store next tile into the alternate buffer
    if (kt < RNN_I / 32 - 1) {
      unsigned short tmp[16];
#pragma unroll
      for (int i = 0; i < 4; ++i) tmp[i]      = f2bf((&areg0.x)[i]);
#pragma unroll
      for (int i = 0; i < 4; ++i) tmp[4 + i]  = f2bf((&areg1.x)[i]);
#pragma unroll
      for (int i = 0; i < 4; ++i) tmp[8 + i]  = f2bf((&areg2.x)[i]);
#pragma unroll
      for (int i = 0; i < 4; ++i) tmp[12 + i] = f2bf((&areg3.x)[i]);
      u16x8* dst = (u16x8*)&As[nxt][arow * 40 + acol];
      dst[0] = *(const u16x8*)&tmp[0];
      dst[1] = *(const u16x8*)&tmp[8];
      *(u16x8*)&Bs[nxt][brow * 40 + bcol] = breg;
    }
    __syncthreads();
  }

  const int mbase = bm * 128 + wm * 32;
  const int nbase = bn * 64 + wn * 32;
  store_ctile_bias(xproj, RNN_H, mbase,      nbase,      acc00, bias);
  store_ctile_bias(xproj, RNN_H, mbase,      nbase + 16, acc01, bias);
  store_ctile_bias(xproj, RNN_H, mbase + 16, nbase,      acc10, bias);
  store_ctile_bias(xproj, RNN_H, mbase + 16, nbase + 16, acc11, bias);
}

// ---------------------------------------------------------------------------
// Kernel 2: persistent recurrent scan. 16 workgroups; WG `wg` owns output
// columns [wg*32, wg*32+32). Its W_hh slice (32x512 bf16, rows padded to 520)
// is staged into LDS once via the Tensor Data Mover (2D tile, data_size=2B,
// LDS pad_interval=256 DWORDs / pad_amount=4 DWORDs -> 520-elem padded rows).
// h ping-pongs in global (bf16) with a generation-counting grid barrier.
__global__ __launch_bounds__(256) void rnn_scan(
    const float* __restrict__ xproj, const unsigned short* __restrict__ Whh,
    unsigned short* __restrict__ hbuf0, unsigned short* __restrict__ hbuf1,
    int* __restrict__ bar) {
  __shared__ unsigned short Ws[32 * 520];   // 512 + 8 pad per row
  __shared__ float Red[2][32][33];

  const int tid = threadIdx.x;
  const int wg  = blockIdx.x;  // 0..15

#if defined(RNN_USE_TDM) && __has_builtin(__builtin_amdgcn_tensor_load_to_lds)
  if ((tid >> 5) == 0) {  // one wave issues the DMA (TDM ignores EXEC)
    const unsigned long long ga =
        (unsigned long long)(uintptr_t)(Whh + (size_t)(wg * 32) * RNN_H);
    const unsigned lds = (unsigned)(uintptr_t)(void*)&Ws[0];  // low 32b = LDS offset
    u32x4 g0 = { 1u,                                   // count=1, user desc
                 lds,                                  // lds_addr
                 (unsigned)(ga & 0xFFFFFFFFu),         // global_addr[31:0]
                 (unsigned)((ga >> 32) & 0x01FFFFFFu) | (2u << 30) };  // addr[56:32] | type=2
    i32x8 g1 = { (int)((1u << 16) | (1u << 20) | (7u << 22) | (3u << 25)),
                 //  data_size=2B   pad_enable  interval=256dw  amount=4dw
                 (int)((RNN_H & 0xFFFFu) << 16),       // tensor_dim0[15:0]
                 (int)((RNN_H >> 16) | ((512u & 0xFFFFu) << 16)),  // dim0 hi | tensor_dim1 lo
                 (int)((512u >> 16) | (RNN_H << 16)),  // dim1 hi | tile_dim0=512
                 (int)32,                              // tile_dim1=32, tile_dim2=0
                 (int)RNN_H,                           // tensor_dim0_stride lo
                 0, 0 };
    i32x4 z4 = { 0, 0, 0, 0 };
    i32x8 z8 = { 0, 0, 0, 0, 0, 0, 0, 0 };
    __builtin_amdgcn_tensor_load_to_lds(g0, g1, z4, z4, z8, 0);
#if __has_builtin(__builtin_amdgcn_s_wait_tensorcnt)
    __builtin_amdgcn_s_wait_tensorcnt(0);
#endif
  }
#else
  {
    const int r  = tid >> 3;          // 0..31
    const int c0 = (tid & 7) << 6;    // 0,64,...,448
    const unsigned short* src = Whh + (size_t)(wg * 32 + r) * RNN_H + c0;
    unsigned short* dst = &Ws[r * 520 + c0];
#pragma unroll
    for (int i = 0; i < 8; ++i)
      *(u16x8*)(dst + i * 8) = *(const u16x8*)(src + i * 8);
  }
#endif
  __syncthreads();

  const int w    = tid >> 5;
  const int kh   = w >> 2;         // K half (0: K 0..255, 1: K 256..511)
  const int mt   = w & 1;          // M tile
  const int nt   = (w >> 1) & 1;   // N tile
  const int lane = tid & 31;

  // epilogue mapping (4 consecutive columns per thread)
  const int ep = tid << 2;
  const int er = ep >> 5;          // batch row 0..31
  const int ec = ep & 31;          // column within slice

  for (int t = 0; t < RNN_S; ++t) {
    const unsigned short* hin = (t & 1) ? hbuf1 : hbuf0;
    unsigned short* hout      = (t & 1) ? hbuf0 : hbuf1;

    f32x8 acc = {};
#pragma unroll
    for (int kk = 0; kk < 8; ++kk) {
      const int kb = kh * 256 + kk * 32;
      bf16x16 a = load_a_frag(hin, RNN_H, mt * 16, kb);   // h_old from L2
      bf16x16 b = load_b_frag(Ws, 520, nt * 16, kb);      // resident weights
      acc = __builtin_amdgcn_wmma_f32_16x16x32_bf16(false, a, false, b, (short)0, acc, false, false);
    }
    {
      const int c  = nt * 16 + (lane & 15);
      const int r0 = mt * 16 + ((lane >> 4) << 3);
#pragma unroll
      for (int v = 0; v < 8; ++v) Red[kh][r0 + v][c] = acc[v];
    }

    // Stream next timestep's xproj slice ahead of the grid barrier
    if (t + 1 < RNN_S)
      __builtin_prefetch(xproj + ((size_t)er * RNN_S + (t + 1)) * RNN_H + wg * 32 + ec, 0, 1);

    __syncthreads();

    // Reduce K-halves, add xproj_t, tanh, emit bf16 h_new (4 elems/thread)
    {
      const float* xp = xproj + ((size_t)er * RNN_S + t) * RNN_H + wg * 32 + ec;
      unsigned short* ho = hout + er * RNN_H + wg * 32 + ec;
#pragma unroll
      for (int i = 0; i < 4; ++i) {
        float v = Red[0][er][ec + i] + Red[1][er][ec + i] + xp[i];
        ho[i] = f2bf(tanhf(v));
      }
    }

    // Grid barrier across the 16 workgroups (monotonic generation counter)
    __threadfence();
    __syncthreads();
    if (tid == 0) {
      __hip_atomic_fetch_add(bar, 1, __ATOMIC_RELEASE, __HIP_MEMORY_SCOPE_AGENT);
      const int target = 16 * (t + 1);
      while (__hip_atomic_load(bar, __ATOMIC_ACQUIRE, __HIP_MEMORY_SCOPE_AGENT) < target)
        __builtin_amdgcn_s_sleep(1);
    }
    __syncthreads();
    __threadfence();
  }
}

// ---------------------------------------------------------------------------
// Kernel 3: out = h_final @ W_fc^T + b_fc  (M=32, N=512, K=512)
// 8 blocks x 8 waves = 64 waves, one 16x16 tile each.
__global__ __launch_bounds__(256) void fc_gemm(
    const unsigned short* __restrict__ hfin, const unsigned short* __restrict__ Wfc,
    const float* __restrict__ bfc, float* __restrict__ out) {
  const int tid = threadIdx.x;
  const int gw = blockIdx.x * 8 + (tid >> 5);  // 0..63
  const int nt = gw & 31;
  const int mt = gw >> 5;
  f32x8 acc = {};
  for (int kt = 0; kt < RNN_H / 32; ++kt) {
    bf16x16 a = load_a_frag(hfin, RNN_H, mt * 16, kt * 32);
    bf16x16 b = load_b_frag(Wfc, RNN_H, nt * 16, kt * 32);
    acc = __builtin_amdgcn_wmma_f32_16x16x32_bf16(false, a, false, b, (short)0, acc, false, false);
  }
  store_ctile_bias(out, RNN_O, mt * 16, nt * 16, acc, bfc);
}

// ---------------------------------------------------------------------------
extern "C" void kernel_launch(void* const* d_in, const int* in_sizes, int n_in,
                              void* d_out, int out_size, void* d_ws, size_t ws_size,
                              hipStream_t stream) {
  const float* x   = (const float*)d_in[0];
  const float* Wxh = (const float*)d_in[1];
  const float* bxh = (const float*)d_in[2];
  const float* Whh = (const float*)d_in[3];
  const float* bhh = (const float*)d_in[4];
  const float* bh  = (const float*)d_in[5];
  const float* Wfc = (const float*)d_in[6];
  const float* bfc = (const float*)d_in[7];
  float* out = (float*)d_out;

  // Workspace layout
  char* ws = (char*)d_ws;
  float*          xproj = (float*)ws;                             // 128 MB f32
  unsigned short* WxhB  = (unsigned short*)(ws + 134217728ull);   // 512 KB bf16
  unsigned short* WhhB  = (unsigned short*)(ws + 134742016ull);   // 512 KB bf16
  unsigned short* WfcB  = (unsigned short*)(ws + 135266304ull);   // 512 KB bf16
  unsigned short* h0    = (unsigned short*)(ws + 135790592ull);   // 32 KB bf16
  unsigned short* h1    = (unsigned short*)(ws + 135856128ull);   // 32 KB bf16
  int*            bar   = (int*)(ws + 135921664ull);
  float*          bsum  = (float*)(ws + 135921920ull);            // 2 KB f32

  init_state<<<64, 256, 0, stream>>>(h0, bar);
  cvt_bf16<<<256, 256, 0, stream>>>(Wxh, WxhB, RNN_H * RNN_I);
  cvt_bf16<<<256, 256, 0, stream>>>(Whh, WhhB, RNN_H * RNN_H);
  cvt_bf16<<<256, 256, 0, stream>>>(Wfc, WfcB, RNN_O * RNN_H);
  bias_prep<<<2, 256, 0, stream>>>(bxh, bhh, bh, bsum);

  xproj_gemm<<<dim3(RNN_H / 64, (RNN_B * RNN_S) / 128), 256, 0, stream>>>(
      x, WxhB, bsum, xproj);

  rnn_scan<<<16, 256, 0, stream>>>(xproj, WhhB, h0, h1, bar);

  // after 2048 steps (last t odd), final h lives in h0
  fc_gemm<<<8, 256, 0, stream>>>(h0, WfcB, bfc, out);
}
